// Decoder_76321568850464
// MI455X (gfx1250) — matrix operand order repaired and checked by reading, not compiled, not run on hardware
//
#include <hip/hip_runtime.h>
#include <hip/hip_bf16.h>
#include <math.h>

// ---------------------------------------------------------------------------
// MI455X (gfx1250) implementation of the windowed-attention + GDFN block.
// Heavy GEMMs run on the matrix pipe via v_wmma_f32_16x16x32_bf16 with both
// operands pre-swizzled into WMMA fragment order:
//   - B (weights) are packed at setup into per-(ntile,ktile) 1KB fragment
//     blocks -> each lane's 16 bf16 are a single contiguous 32B global load.
//   - A is staged once per workgroup into LDS in fragment order -> each
//     lane's fragment is a contiguous 32B LDS load; no barriers in K-loop.
// ---------------------------------------------------------------------------

typedef __attribute__((ext_vector_type(16))) __bf16 v16bf;
typedef __attribute__((ext_vector_type(8)))  float  v8f;

#define IMG    176
#define HW     30976          // 176*176
#define CD     128
#define NPIX   123904         // 4*176*176  (== total tokens: 4*44*44*16)
#define NWIN   44
#define LNEPS  1e-5f

// ---------------------------------------------------------------------------
// Swizzled-B element decode: flat fragment-order index -> (k, n_local).
// Layout: idx = ((nt*(K/32) + kt)*32 + lane)*16 + e
//         lane = col + 16*half ; k = kt*32 + (e<8 ? 8*half+e : 16+8*half+(e-8)) ; n = nt*16 + col
// This matches the CDNA5 ISA 7.12.2 16-bit B 32x16 per-lane layout, so the
// GEMM can load each lane's B fragment as 32 contiguous bytes.
// ---------------------------------------------------------------------------
__device__ __forceinline__ void bsw_decode(int idx, int ktiles,
                                           int& k, int& n) {
  int e    = idx & 15;
  int lane = (idx >> 4) & 31;
  int t2   = idx >> 9;
  int kt   = t2 % ktiles;
  int nt   = t2 / ktiles;
  k = kt * 32 + (e < 8 ? (lane >> 4) * 8 + e : 16 + (lane >> 4) * 8 + (e - 8));
  n = nt * 16 + (lane & 15);
}

// ---------------------------------------------------------------------------
// Weight packing: fp32 -> bf16 fragment-order blocks.
//   Bq/Bk/Bv : K=128, N=128            from W[o=n][c=k]
//   Bin      : 4 chunks, each K=128, N=512 (cols = 256 gate-in || 256 gate-mul)
//   Bout     : 4 chunks, each K=256, N=128 from w_out[o=n][hid]
// ---------------------------------------------------------------------------
__global__ __launch_bounds__(256) void pack_weights(
    const float* __restrict__ Wq, const float* __restrict__ Wk,
    const float* __restrict__ Wv, const float* __restrict__ w_in,
    const float* __restrict__ w_out,
    __bf16* __restrict__ Bq, __bf16* __restrict__ Bk, __bf16* __restrict__ Bv,
    __bf16* __restrict__ Bin, __bf16* __restrict__ Bout)
{
  int idx = blockIdx.x * 256 + threadIdx.x;
  int k, n;
  if (idx < 16384) { bsw_decode(idx, 4, k, n); Bq[idx] = (__bf16)Wq[n*CD + k]; return; }
  idx -= 16384;
  if (idx < 16384) { bsw_decode(idx, 4, k, n); Bk[idx] = (__bf16)Wk[n*CD + k]; return; }
  idx -= 16384;
  if (idx < 16384) { bsw_decode(idx, 4, k, n); Bv[idx] = (__bf16)Wv[n*CD + k]; return; }
  idx -= 16384;
  if (idx < 262144) {                       // Bin: chunk stride 128*512
    int ch = idx >> 16, li = idx & 65535;
    bsw_decode(li, 4, k, n);                // k<128, n<512
    int o = (n < 256) ? (ch*256 + n) : (1024 + ch*256 + (n - 256));
    Bin[idx] = (__bf16)w_in[o*CD + k];
    return;
  }
  idx -= 262144;
  if (idx < 131072) {                       // Bout: chunk stride 256*128
    int ch = idx >> 15, li = idx & 32767;
    bsw_decode(li, 8, k, n);                // k<256, n<128
    Bout[idx] = (__bf16)w_out[n*512 + (ch*256 + k)];
  }
}

// ---------------------------------------------------------------------------
// Bilinear resize helper (jax.image.resize: half-pixel centers, edge clamp).
// Fixed 44 -> 176 upsample: src = (dst + 0.5)*0.25 - 0.5
// ---------------------------------------------------------------------------
__device__ __forceinline__ void bilin_setup(int o, int& i0, int& i1, float& w) {
  float s = ((float)o + 0.5f) * 0.25f - 0.5f;
  float f = floorf(s);
  int a = (int)f;
  w = s - f;
  i0 = min(NWIN - 1, max(0, a));
  i1 = min(NWIN - 1, max(0, a + 1));
}

// ---------------------------------------------------------------------------
// LayerNorm-1 over channels + mask/edge bilinear resize, packed to bf16 GEMM
// A-operands (pixel-major rows of 128):
//   Ae[p][c] = edge_resized            (feeds Q GEMM)
//   Ax[p][c] = LN(x)*mask_resized      (feeds K/V GEMMs)
// ---------------------------------------------------------------------------
__global__ __launch_bounds__(256) void ln1_resize_pack(
    const float* __restrict__ x, const float* __restrict__ mask,
    const float* __restrict__ edge, const float* __restrict__ gma,
    const float* __restrict__ bta, __bf16* __restrict__ Ae,
    __bf16* __restrict__ Ax)
{
  int p = blockIdx.x * 256 + threadIdx.x;            // < NPIX
  int b = p / HW, rem = p % HW, y = rem / IMG, xx = rem % IMG;
  const float* xb = x + (size_t)b * CD * HW + rem;   // (b,c,y,x) at xb[c*HW]
  float s = 0.f, s2 = 0.f;
  for (int c = 0; c < CD; ++c) { float v = xb[(size_t)c * HW]; s += v; s2 += v * v; }
  float mu  = s * (1.f / CD);
  float var = s2 * (1.f / CD) - mu * mu;
  float inv = rsqrtf(var + LNEPS);

  int y0, y1, x0, x1; float wy, wx;
  bilin_setup(y, y0, y1, wy);
  bilin_setup(xx, x0, x1, wx);
  const float* mb = mask + b * NWIN * NWIN;
  float mval = (1.f - wy) * ((1.f - wx) * mb[y0*NWIN + x0] + wx * mb[y0*NWIN + x1])
             +        wy  * ((1.f - wx) * mb[y1*NWIN + x0] + wx * mb[y1*NWIN + x1]);

  const float* eb = edge + (size_t)b * CD * NWIN * NWIN;
  __bf16* ae = Ae + (size_t)p * CD;
  __bf16* ax = Ax + (size_t)p * CD;
  for (int c = 0; c < CD; ++c) {
    float v  = xb[(size_t)c * HW];
    float xn = (v - mu) * inv * gma[c] + bta[c];
    ax[c] = (__bf16)(xn * mval);
    const float* ec = eb + (size_t)c * NWIN * NWIN;
    float ev = (1.f - wy) * ((1.f - wx) * ec[y0*NWIN + x0] + wx * ec[y0*NWIN + x1])
             +        wy  * ((1.f - wx) * ec[y1*NWIN + x0] + wx * ec[y1*NWIN + x1]);
    ae[c] = (__bf16)ev;
  }
}

// ---------------------------------------------------------------------------
// WMMA GEMM:  C[M x N] (+)= A[M x K](bf16, row-major) * Bsw[K x N](bf16,
// fragment-order). Output f32 (store_bf16=0; accumulate supported) or bf16.
// Block = 256 threads = 8 wave32s, tile 16(M) x 128(N), one 16x16 C tile per
// wave. The full 16xK A block is staged once into LDS *in fragment order*
// (loader computes the ISA A 16x32 per-lane layout), so the K-loop is just:
//   2x ds_load_b128 (A frag) + 2x global_load_b128 (B frag) + 1x v_wmma
// with no barriers. Grid exact (M%16==0, N%128==0): EXEC all-ones at WMMA.
// ---------------------------------------------------------------------------
__global__ __launch_bounds__(256) void gemm_bf16_wmma(
    const __bf16* __restrict__ A, const __bf16* __restrict__ Bsw,
    void* __restrict__ Cv, int Kdim, int ldC, int accumulate, int store_bf16)
{
  __shared__ __align__(16) __bf16 sA[16 * 512];   // fragment-order A block

  float*  Cf = (float*)Cv;
  __bf16* Ch = (__bf16*)Cv;

  const int tid  = threadIdx.x;
  const int lane = tid & 31, wv = tid >> 5;
  const int row  = lane & 15, half = lane >> 4;
  const int m0   = blockIdx.x * 16;
  const int nt16 = blockIdx.y * 8 + wv;           // global 16-col tile index
  const int ktiles = Kdim >> 5;

  // ---- stage A[m0..m0+15][0..K) into LDS in WMMA fragment order ----------
  // src dword d = (row, kp): holds A[row][2kp..2kp+1].
  // dest: kt = kp>>4; kpl = kp&15;
  //   kpl<8 : half=kpl>>2, slot=kpl&3        (K-local 8*half+2p)
  //   kpl>=8: half=(kpl-8)>>2, slot=4+((kpl-8)&3)  (K-local 16+8*half+2p)
  // frag dword address = (kt*32 + row + 16*half)*8 + slot
  const int halfK = Kdim >> 1;
  for (int d = tid; d < Kdim * 8; d += 256) {
    int r  = d / halfK, kp = d % halfK;
    unsigned int val =
        *(const unsigned int*)(A + (size_t)(m0 + r) * Kdim + 2 * kp);
    int kt = kp >> 4, kpl = kp & 15;
    int hf, slot;
    if (kpl < 8) { hf = kpl >> 2; slot = kpl & 3; }
    else         { hf = (kpl - 8) >> 2; slot = 4 + ((kpl - 8) & 3); }
    ((unsigned int*)sA)[(kt * 32 + r + 16 * hf) * 8 + slot] = val;
  }

  v8f accf;
  if (accumulate) {
#pragma unroll
    for (int r = 0; r < 8; ++r)
      accf[r] = Cf[(size_t)(m0 + r + 8 * half) * ldC + nt16 * 16 + row];
  } else {
#pragma unroll
    for (int r = 0; r < 8; ++r) accf[r] = 0.f;
  }

  __syncthreads();                                // A block ready (only barrier)

  // ---- K loop: pure fragment loads + wmma --------------------------------
  const __bf16* bptr = Bsw + ((size_t)nt16 * ktiles * 32 + lane) * 16;
  const __bf16* aptr = sA + lane * 16;
#pragma unroll 4
  for (int kt = 0; kt < ktiles; ++kt) {
    v16bf afr = *(const v16bf*)(aptr + kt * 512);   // 2x ds_load_b128
    v16bf bfr = *(const v16bf*)(bptr + kt * 512);   // 2x global_load_b128
    accf = __builtin_amdgcn_wmma_f32_16x16x32_bf16(
        false, afr, false, bfr, (short)0, accf, false, false);
  }

  if (store_bf16) {
#pragma unroll
    for (int r = 0; r < 8; ++r)
      Ch[(size_t)(m0 + r + 8 * half) * ldC + nt16 * 16 + row] = (__bf16)accf[r];
  } else {
#pragma unroll
    for (int r = 0; r < 8; ++r)
      Cf[(size_t)(m0 + r + 8 * half) * ldC + nt16 * 16 + row] = accf[r];
  }
}

// ---------------------------------------------------------------------------
// Per-token 16x16 channel attention + faithful scrambled window-merge +
// residual. One wave32 per token (8 tokens/block). The merge is bijective:
//   X1[b][(n*128+cc)*1936 + wh*44 + ww] = x[same] + out(token, cc)
// ---------------------------------------------------------------------------
__global__ __launch_bounds__(256) void attn_scatter(
    const float* __restrict__ Q, const float* __restrict__ Kt,
    const float* __restrict__ Vt, const float* __restrict__ xin,
    float* __restrict__ X1)
{
  __shared__ __align__(16) float sq[8][128];
  __shared__ __align__(16) float sk[8][128];
  __shared__ __align__(16) float sv[8][128];
  __shared__ __align__(16) float sat[8][256];
  const int lane = threadIdx.x & 31, wv = threadIdx.x >> 5;

  int t = blockIdx.x * 8 + wv;                 // token id, < NPIX
  int n = t & 15; int u = t >> 4;
  int ww = u % NWIN; u /= NWIN;
  int wh = u % NWIN; int b = u / NWIN;
  int y  = wh * 4 + (n >> 2), xx = ww * 4 + (n & 3);
  int p  = (b * IMG + y) * IMG + xx;           // source pixel of this token

  ((float4*)sq[wv])[lane] = ((const float4*)(Q  + (size_t)p * CD))[lane];
  ((float4*)sk[wv])[lane] = ((const float4*)(Kt + (size_t)p * CD))[lane];
  ((float4*)sv[wv])[lane] = ((const float4*)(Vt + (size_t)p * CD))[lane];
  __syncthreads();

  // dots[j][k] = scale * sum_i q[i,j]*k[i,k]   (contract over heads i=0..7)
#pragma unroll
  for (int m = 0; m < 8; ++m) {
    int idx = lane * 8 + m, j = idx >> 4, kk = idx & 15;
    float s = 0.f;
#pragma unroll
    for (int i = 0; i < 8; ++i) s += sq[wv][i * 16 + j] * sk[wv][i * 16 + kk];
    sat[wv][idx] = s * 0.25f;                  // scale = 16^-0.5
  }
  __syncthreads();

  if (lane < 16) {                             // row-wise softmax
    float mx = -3.0e38f, ev[16], sum = 0.f;
#pragma unroll
    for (int kk = 0; kk < 16; ++kk) mx = fmaxf(mx, sat[wv][lane * 16 + kk]);
#pragma unroll
    for (int kk = 0; kk < 16; ++kk) { ev[kk] = __expf(sat[wv][lane * 16 + kk] - mx); sum += ev[kk]; }
    float invs = 1.f / sum;
#pragma unroll
    for (int kk = 0; kk < 16; ++kk) sat[wv][lane * 16 + kk] = ev[kk] * invs;
  }
  __syncthreads();

  // out[i,j] = sum_k attn[j,k] * v[i,k]; scatter-merge + residual
  size_t base = (size_t)b * (CD * HW);
#pragma unroll
  for (int m = 0; m < 4; ++m) {
    int cc = lane * 4 + m, i = cc >> 4, j = cc & 15;
    float o = 0.f;
#pragma unroll
    for (int kk = 0; kk < 16; ++kk) o += sv[wv][i * 16 + kk] * sat[wv][j * 16 + kk];
    size_t f = (size_t)(n * CD + cc) * (NWIN * NWIN) + wh * NWIN + ww;
    X1[base + f] = xin[base + f] + o;
  }
}

// ---------------------------------------------------------------------------
// LayerNorm-2 over channels of X1 -> bf16 A-operand (pixel-major).
// ---------------------------------------------------------------------------
__global__ __launch_bounds__(256) void ln2_pack(
    const float* __restrict__ X1, const float* __restrict__ gma,
    const float* __restrict__ bta, __bf16* __restrict__ A2)
{
  int p = blockIdx.x * 256 + threadIdx.x;
  int b = p / HW, rem = p % HW;
  const float* xb = X1 + (size_t)b * CD * HW + rem;
  float s = 0.f, s2 = 0.f;
  for (int c = 0; c < CD; ++c) { float v = xb[(size_t)c * HW]; s += v; s2 += v * v; }
  float mu  = s * (1.f / CD);
  float var = s2 * (1.f / CD) - mu * mu;
  float inv = rsqrtf(var + LNEPS);
  __bf16* a = A2 + (size_t)p * CD;
  for (int c = 0; c < CD; ++c) {
    float v = xb[(size_t)c * HW];
    a[c] = (__bf16)((v - mu) * inv * gma[c] + bta[c]);
  }
}

// ---------------------------------------------------------------------------
// Depthwise 3x3 (pad 1) + exact GELU gate for one 512-channel chunk:
// Ychunk[p][0..255]=gate-in channels, [256..511]=gate-mul channels.
// G[p][i] = gelu(dw(y1_i)) * dw(y2_i), bf16 (GEMM-2 A operand).
// ---------------------------------------------------------------------------
__global__ __launch_bounds__(256) void dwconv_gelu(
    const __bf16* __restrict__ Y, const float* __restrict__ wdw,
    __bf16* __restrict__ G, int ch)
{
  int idx = blockIdx.x * 256 + threadIdx.x;    // < NPIX*256
  int i = idx & 255; int p = idx >> 8;
  int rem = p % HW, y = rem / IMG, xx = rem % IMG;
  int o1 = ch * 256 + i, o2 = 1024 + ch * 256 + i;
  const float* w1 = wdw + o1 * 9;
  const float* w2 = wdw + o2 * 9;
  float a1 = 0.f, a2 = 0.f;
#pragma unroll
  for (int dy = -1; dy <= 1; ++dy) {
    int yy = y + dy; if (yy < 0 || yy >= IMG) continue;
#pragma unroll
    for (int dx = -1; dx <= 1; ++dx) {
      int xc = xx + dx; if (xc < 0 || xc >= IMG) continue;
      const __bf16* yp = Y + (size_t)(p + dy * IMG + dx) * 512;
      int tap = (dy + 1) * 3 + (dx + 1);
      a1 += w1[tap] * (float)yp[i];
      a2 += w2[tap] * (float)yp[256 + i];
    }
  }
  float gl = 0.5f * a1 * (1.f + erff(a1 * 0.70710678118654752f));
  G[(size_t)p * 256 + i] = (__bf16)(gl * a2);
}

// ---------------------------------------------------------------------------
// Final residual: out(b,c,y,x) = X1(b,c,y,x) + C2[pixel][c]
// ---------------------------------------------------------------------------
__global__ __launch_bounds__(256) void residual_out(
    const float* __restrict__ X1, const float* __restrict__ C2,
    float* __restrict__ out)
{
  int idx = blockIdx.x * 256 + threadIdx.x;    // < NPIX*CD
  int b = idx / (CD * HW); int f = idx % (CD * HW);
  int c = f / HW; int rem = f % HW;
  int p = b * HW + rem;
  out[idx] = X1[idx] + C2[(size_t)p * CD + c];
}

// ---------------------------------------------------------------------------
extern "C" void kernel_launch(void* const* d_in, const int* in_sizes, int n_in,
                              void* d_out, int out_size, void* d_ws, size_t ws_size,
                              hipStream_t stream)
{
  (void)in_sizes; (void)n_in; (void)out_size; (void)ws_size;
  const float* x    = (const float*)d_in[0];
  const float* mask = (const float*)d_in[1];
  const float* edge = (const float*)d_in[2];
  const float* ln1w = (const float*)d_in[3];
  const float* ln1b = (const float*)d_in[4];
  const float* Wq   = (const float*)d_in[5];
  const float* Wk   = (const float*)d_in[6];
  const float* Wv   = (const float*)d_in[7];
  const float* ln2w = (const float*)d_in[8];
  const float* ln2b = (const float*)d_in[9];
  const float* win  = (const float*)d_in[10];
  const float* wdw  = (const float*)d_in[11];
  const float* wout = (const float*)d_in[12];

  char* ws = (char*)d_ws;
  const size_t szF = (size_t)NPIX * CD * 4;    // 63,438,848 B  (P x 128 f32)
  const size_t szH = (size_t)NPIX * CD * 2;    // 31,719,424 B  (P x 128 bf16)

  float*  X1  = (float*)(ws);                          // residual after attn
  __bf16* Ae  = (__bf16*)(ws + szF);                   // edge_r (Q GEMM A)
  __bf16* Ax  = (__bf16*)(ws + szF + szH);             // LN(x)*mask (K/V A)
  float*  Qb  = (float*)(ws + szF + 2 * szH);
  float*  Kb  = (float*)(ws + 2 * szF + 2 * szH);
  float*  Vb  = (float*)(ws + 3 * szF + 2 * szH);
  char*   wsw = ws + 4 * szF + 2 * szH;
  __bf16* Bq   = (__bf16*)wsw;                         // 128x128 frag-order
  __bf16* Bk   = Bq + 16384;
  __bf16* Bv   = Bk + 16384;
  __bf16* Bin  = Bv + 16384;                           // 4 x (128x512)
  __bf16* Bout = Bin + 262144;                         // 4 x (256x128)
  float*  C2   = (float*)(wsw + 2 * (3*16384 + 262144 + 131072)); // GDFN accum
  // Phase-B reuse (Q/K/V dead after attention): chunk activations.
  __bf16* Ych = (__bf16*)Qb;                           // P x 512 bf16 (=2*szF)
  __bf16* Gb  = (__bf16*)Vb;                           // P x 256 bf16 (=szF)
  __bf16* A2  = Ae;                                    // Ae dead after Q GEMM

  // total pack elements = 3*16384 + 262144 + 131072 = 442368 = 1728 * 256
  pack_weights<<<1728, 256, 0, stream>>>(Wq, Wk, Wv, win, wout, Bq, Bk, Bv, Bin, Bout);
  ln1_resize_pack<<<NPIX / 256, 256, 0, stream>>>(x, mask, edge, ln1w, ln1b, Ae, Ax);

  dim3 g1(NPIX / 16, 1);                               // M=123904, N=128
  gemm_bf16_wmma<<<g1, 256, 0, stream>>>(Ae, Bq, (void*)Qb, 128, 128, 0, 0);
  gemm_bf16_wmma<<<g1, 256, 0, stream>>>(Ax, Bk, (void*)Kb, 128, 128, 0, 0);
  gemm_bf16_wmma<<<g1, 256, 0, stream>>>(Ax, Bv, (void*)Vb, 128, 128, 0, 0);

  attn_scatter<<<NPIX / 8, 256, 0, stream>>>(Qb, Kb, Vb, x, X1);
  ln2_pack<<<NPIX / 256, 256, 0, stream>>>(X1, ln2w, ln2b, A2);

  for (int ch = 0; ch < 4; ++ch) {
    dim3 g2(NPIX / 16, 4);                             // N=512 per chunk
    gemm_bf16_wmma<<<g2, 256, 0, stream>>>(A2, Bin + ch * 65536, (void*)Ych,
                                           128, 512, 0, 1);
    dwconv_gelu<<<NPIX, 256, 0, stream>>>(Ych, wdw, Gb, ch);
    gemm_bf16_wmma<<<g1, 256, 0, stream>>>(Gb, Bout + ch * 32768, (void*)C2,
                                           256, 128, ch ? 1 : 0, 0);
  }

  residual_out<<<(NPIX * CD) / 256, 256, 0, stream>>>(X1, C2, (float*)d_out);
}
